// LinearAttentionBlock_25563645346554
// MI455X (gfx1250) — matrix operand order repaired and checked by reading, not compiled, not run on hardware
//
#include <hip/hip_runtime.h>
#include <hip/hip_bf16.h>

// ---------------- types / helpers ----------------
typedef __bf16 bf16_t;
typedef bf16_t v16bf __attribute__((ext_vector_type(16)));
typedef float  v8f   __attribute__((ext_vector_type(8)));

struct Pack8 { unsigned u[8]; };
struct alignas(16) Frag32 { uint4 lo, hi; };   // 32 bytes = one lane's v16bf

__device__ __forceinline__ unsigned short f2bf(float f) {
    union { float f; unsigned u; } v; v.f = f;
    unsigned r = v.u + 0x7FFFu + ((v.u >> 16) & 1u);
    return (unsigned short)(r >> 16);
}
__device__ __forceinline__ float elu1(float x) {
    return x > 0.0f ? x + 1.0f : __expf(x);
}
__device__ __forceinline__ v8f wmma_bf16(v16bf a, v16bf b, v8f c) {
    return __builtin_amdgcn_wmma_f32_16x16x32_bf16(false, a, false, b,
                                                   (short)0, c, false, false);
}

// load one 32-byte lane-fragment from a fragment-tiled global buffer
__device__ __forceinline__ v16bf ldfragg(const unsigned short* g, long elemOff) {
    Frag32 f = *(const Frag32*)(g + elemOff);
    return __builtin_bit_cast(v16bf, f);
}

// half/e decomposition for the A-fragment K index
__device__ __forceinline__ void a_halfe(int kk, int& half, int& e) {
    if (kk < 16) { half = kk >> 3; e = kk & 7; }
    else         { int k2 = kk - 16; half = k2 >> 3; e = (k2 & 7) + 8; }
}
// full producer offsets (used only in low-rate reformat kernels)
__device__ __forceinline__ long bfrag_off(int k, int n, int kt32) {
    int kk = k & 31;
    int half = (kk >> 3) & 1, e = (kk & 7) + ((kk >> 4) << 3);
    long frag = (long)(n >> 4) * kt32 + (k >> 5);
    return frag * 512 + (long)(half * 16 + (n & 15)) * 16 + e;
}

// A fragment from LDS (row-major tile, stride lda)
__device__ __forceinline__ v16bf load_frag_a(const unsigned short* lds, int row0,
                                             int lda, int lane) {
    int half = lane >> 4;
    const unsigned short* p = lds + (row0 + (lane & 15)) * lda;
    Pack8 t;
#pragma unroll
    for (int i = 0; i < 4; ++i) t.u[i] = *(const unsigned*)(p + half * 8 + 2 * i);
#pragma unroll
    for (int i = 0; i < 4; ++i) t.u[4 + i] = *(const unsigned*)(p + 16 + half * 8 + 2 * i);
    return __builtin_bit_cast(v16bf, t);
}

// ---------------- constants ----------------
#define D_MODEL 768
#define N_HEADS 8
#define HEAD_DIM 96
#define SEQ_L 4096
#define BATCH 8
#define ROWS_TOTAL (BATCH * SEQ_L)   // 32768
#define BH (BATCH * N_HEADS)         // 64
#define VPAD 128
#define KSPLIT 8
#define QF_PER_BH  ((long)SEQ_L * HEAD_DIM)        // 393216 (A-layout, kt32=3)
#define KT_PER_BH  ((long)HEAD_DIM * SEQ_L)        // 393216 (A-layout, kt32=128)
#define VP_PER_BH  ((long)SEQ_L * VPAD)            // 524288 (B-layout, kt32=128)
#define KVE_PER_BH ((long)HEAD_DIM * VPAD)         // 12288  (B-layout, kt32=3)

// ---------------- K0: weights fp32 -> bf16, fragment-tiled ----------------
__global__ __launch_bounds__(256) void la_wconv(const float* __restrict__ wqkv,
                                                const float* __restrict__ wout,
                                                unsigned short* __restrict__ wqkv_bf,
                                                unsigned short* __restrict__ wout_bf) {
    long i = (long)blockIdx.x * 256 + threadIdx.x;
    const long nq = (long)D_MODEL * 3 * D_MODEL;
    const long no = (long)D_MODEL * D_MODEL;
    if (i < nq) {
        int k = (int)(i / (3 * D_MODEL)), n = (int)(i % (3 * D_MODEL));
        wqkv_bf[bfrag_off(k, n, 24)] = f2bf(wqkv[i]);
    } else if (i < nq + no) {
        long j = i - nq;
        int k = (int)(j / D_MODEL), n = (int)(j % D_MODEL);
        wout_bf[bfrag_off(k, n, 24)] = f2bf(wout[j]);
    }
}

// ---------------- K1: LayerNorm -> bf16 row-major (wave per row) ----------------
__global__ __launch_bounds__(256) void la_layernorm(const float* __restrict__ x,
                                                    const float* __restrict__ gamma,
                                                    const float* __restrict__ beta,
                                                    unsigned short* __restrict__ xn) {
    int w = threadIdx.x >> 5, lane = threadIdx.x & 31;
    long row = (long)blockIdx.x * 8 + w;
    const float* px = x + row * D_MODEL;
    float v[24];
    float s = 0.0f;
#pragma unroll
    for (int i = 0; i < 24; ++i) { v[i] = px[lane + 32 * i]; s += v[i]; }
#pragma unroll
    for (int o = 16; o > 0; o >>= 1) s += __shfl_xor(s, o);
    float mu = s * (1.0f / 768.0f);
    float s2 = 0.0f;
#pragma unroll
    for (int i = 0; i < 24; ++i) { float d = v[i] - mu; s2 += d * d; }
#pragma unroll
    for (int o = 16; o > 0; o >>= 1) s2 += __shfl_xor(s2, o);
    float rs = rsqrtf(s2 * (1.0f / 768.0f) + 1e-5f);
    unsigned short* po = xn + row * D_MODEL;
#pragma unroll
    for (int i = 0; i < 24; ++i) {
        int c = lane + 32 * i;
        po[c] = f2bf((v[i] - mu) * rs * gamma[c] + beta[c]);
    }
}

// ---------------- K2: QKV GEMM; A via double-buffered LDS, B frag-direct ----------------
__device__ __forceinline__ void k2_load_regs(const unsigned short* g, int kcol,
                                             int tid, uint4& a0, uint4& a1) {
    int r0 = tid >> 2, c0 = (tid & 3) << 3;
    a0 = *(const uint4*)(g + (long)r0 * D_MODEL + kcol + c0);
    a1 = *(const uint4*)(g + (long)(r0 + 64) * D_MODEL + kcol + c0);
}
__device__ __forceinline__ void k2_store_lds(unsigned short* lds, int tid,
                                             uint4 a0, uint4 a1) {
    int r0 = tid >> 2, c0 = (tid & 3) << 3;
    *(uint4*)(lds + r0 * 40 + c0) = a0;
    *(uint4*)(lds + (r0 + 64) * 40 + c0) = a1;
}

__global__ __launch_bounds__(256) void la_gemm_qkv(const unsigned short* __restrict__ xn,
                                                   const unsigned short* __restrict__ wqf,
                                                   unsigned short* __restrict__ qff,
                                                   unsigned short* __restrict__ ktf,
                                                   unsigned short* __restrict__ vpf) {
    __shared__ unsigned short la[2][128 * 40];
    int tid = threadIdx.x, lane = tid & 31, w = tid >> 5;
    int wy = w >> 2, wx = w & 3;               // 2x4 waves, wave tile 64x32
    int m0 = blockIdx.y * 128, n0 = blockIdx.x * 128;
    const unsigned short* gA = xn + (long)m0 * D_MODEL;
    v8f acc[4][2];
    v8f z = {0.f, 0.f, 0.f, 0.f, 0.f, 0.f, 0.f, 0.f};
#pragma unroll
    for (int i = 0; i < 4; ++i)
#pragma unroll
        for (int j = 0; j < 2; ++j) acc[i][j] = z;

    uint4 pa0, pa1;
    k2_load_regs(gA, 0, tid, pa0, pa1);
    k2_store_lds(la[0], tid, pa0, pa1);

    for (int kt = 0; kt < 24; ++kt) {
        __syncthreads();
        int cur = kt & 1;
        bool more = (kt + 1 < 24);
        if (more) k2_load_regs(gA, (kt + 1) * 32, tid, pa0, pa1);
        if (kt + 2 < 24)
            __builtin_prefetch(gA + (long)(tid >> 2) * D_MODEL + (kt + 2) * 32 +
                                   ((tid & 3) << 3), 0, 1);
        v16bf bfr[2];
#pragma unroll
        for (int fn = 0; fn < 2; ++fn) {
            int nt = (n0 >> 4) + wx * 2 + fn;
            bfr[fn] = ldfragg(wqf, ((long)nt * 24 + kt) * 512 + lane * 16);
        }
#pragma unroll
        for (int fm = 0; fm < 4; ++fm) {
            v16bf af = load_frag_a(la[cur], wy * 64 + fm * 16, 40, lane);
#pragma unroll
            for (int fn = 0; fn < 2; ++fn) acc[fm][fn] = wmma_bf16(af, bfr[fn], acc[fm][fn]);
        }
        if (more) k2_store_lds(la[cur ^ 1], tid, pa0, pa1);
    }

    // ---- epilogue: hoisted fragment addressing, per-(fm,fn) bases ----
    int hi8 = (lane >> 4) * 8;
    int mbase16 = (m0 >> 4) + wy * 4;          // m>>4 = mbase16 + fm
#pragma unroll
    for (int fn = 0; fn < 2; ++fn) {
        int n = n0 + wx * 32 + fn * 16 + (lane & 15);
        if (n < D_MODEL) {                     // ---- Q: A-layout (l, dd), kt32=3
            int h = n / HEAD_DIM, dd = n - h * HEAD_DIM;
            int half, e;
            a_halfe(dd & 31, half, e);
#pragma unroll
            for (int fm = 0; fm < 4; ++fm) {
                int m16 = mbase16 + fm;
                int b = m16 >> 8, l16 = m16 & 255;    // l>>4
                long base = (long)(b * N_HEADS + h) * QF_PER_BH +
                            ((long)l16 * 3 + (dd >> 5)) * 512 +
                            (long)(half * 16 + hi8) * 16 + e;
#pragma unroll
                for (int r = 0; r < 8; ++r)
                    qff[base + r * 16] = f2bf(elu1(acc[fm][fn][r]));
            }
        } else if (n < 2 * D_MODEL) {          // ---- K^T: A-layout (dd, l), kt32=128
            int n2 = n - D_MODEL;
            int h = n2 / HEAD_DIM, dd = n2 - h * HEAD_DIM;
#pragma unroll
            for (int fm = 0; fm < 4; ++fm) {
                int m16 = mbase16 + fm;
                int b = m16 >> 8;
                int lb = (m16 & 255) << 4;     // 16-aligned l base (l = lb + r + hi8)
                int e0 = (lb & 16) ? 8 : 0;
                long base = (long)(b * N_HEADS + h) * KT_PER_BH +
                            ((long)(dd >> 4) * 128 + (lb >> 5)) * 512 +
                            (long)(dd & 15) * 16 + e0;
#pragma unroll
                for (int r = 0; r < 8; ++r) {
                    int rr = r + hi8;
                    ktf[base + ((rr >> 3) << 8) + (rr & 7)] = f2bf(elu1(acc[fm][fn][r]));
                }
            }
        } else {                               // ---- V: B-layout (l, dd), kt32=128
            int n2 = n - 2 * D_MODEL;
            int h = n2 / HEAD_DIM, dd = n2 - h * HEAD_DIM;
#pragma unroll
            for (int fm = 0; fm < 4; ++fm) {
                int m16 = mbase16 + fm;
                int b = m16 >> 8;
                int lb = (m16 & 255) << 4;
                int e0 = (lb & 16) ? 8 : 0;    // half = hi8/8, e = e0 + r (table)
                long bhbase = (long)(b * N_HEADS + h) * VP_PER_BH;
                long base = bhbase +
                            ((long)(dd >> 4) * 128 + (lb >> 5)) * 512 +
                            (long)((hi8 >> 3) * 16 + (dd & 15)) * 16 + e0;
                unsigned pk[4];
#pragma unroll
                for (int i = 0; i < 4; ++i)
                    pk[i] = (unsigned)f2bf(acc[fm][fn][2 * i]) |
                            ((unsigned)f2bf(acc[fm][fn][2 * i + 1]) << 16);
                uint4 sv = {pk[0], pk[1], pk[2], pk[3]};
                *(uint4*)(vpf + base) = sv;                 // one b128 store
                if (dd < 32) {                 // pad cols 96..127 (col 96 = ones)
                    long base2 = bhbase +
                                 ((long)(6 + (dd >> 4)) * 128 + (lb >> 5)) * 512 +
                                 (long)((hi8 >> 3) * 16 + (dd & 15)) * 16 + e0;
                    unsigned pp = (dd == 0)
                                      ? ((unsigned)f2bf(1.0f) | ((unsigned)f2bf(1.0f) << 16))
                                      : 0u;
                    uint4 sv2 = {pp, pp, pp, pp};
                    *(uint4*)(vpf + base2) = sv2;
                }
            }
        }
    }
}

// ---------------- K3a: KV aggregation split-K, LDS-free ----------------
__global__ __launch_bounds__(256) void la_kv_part(const unsigned short* __restrict__ ktf,
                                                  const unsigned short* __restrict__ vpf,
                                                  float* __restrict__ kvp) {
    int tid = threadIdx.x, lane = tid & 31, w = tid >> 5;
    int wy = w >> 2, wx = w & 3;               // wave tile 48x32 (3x2 frags)
    int ks = blockIdx.x, bh = blockIdx.y;
    const unsigned short* A  = ktf + (long)bh * KT_PER_BH;
    const unsigned short* Bv = vpf + (long)bh * VP_PER_BH;
    v8f acc[3][2];
    v8f z = {0.f, 0.f, 0.f, 0.f, 0.f, 0.f, 0.f, 0.f};
#pragma unroll
    for (int i = 0; i < 3; ++i)
#pragma unroll
        for (int j = 0; j < 2; ++j) acc[i][j] = z;

    for (int i = 0; i < 16; ++i) {
        int lt = ks * 16 + i;
        v16bf bfr[2];
#pragma unroll
        for (int fn = 0; fn < 2; ++fn) {
            int nt = wx * 2 + fn;
            bfr[fn] = ldfragg(Bv, ((long)nt * 128 + lt) * 512 + lane * 16);
        }
        if (i + 1 < 16)
            __builtin_prefetch(A + ((long)(wy * 3) * 128 + lt + 1) * 512 + lane * 16, 0, 1);
#pragma unroll
        for (int fm = 0; fm < 3; ++fm) {
            int mt = wy * 3 + fm;
            v16bf af = ldfragg(A, ((long)mt * 128 + lt) * 512 + lane * 16);
#pragma unroll
            for (int fn = 0; fn < 2; ++fn) acc[fm][fn] = wmma_bf16(af, bfr[fn], acc[fm][fn]);
        }
    }
    // n-major partial store: two float4 per (fm,fn) per lane
    int hi8 = (lane >> 4) * 8;
    float* out = kvp + ((long)(bh * KSPLIT + ks)) * VPAD * HEAD_DIM;
#pragma unroll
    for (int fn = 0; fn < 2; ++fn) {
        int n = wx * 32 + fn * 16 + (lane & 15);
#pragma unroll
        for (int fm = 0; fm < 3; ++fm) {
            int m = wy * 48 + fm * 16 + hi8;
            float* p = out + (long)n * HEAD_DIM + m;
            float4 s0 = {acc[fm][fn][0], acc[fm][fn][1], acc[fm][fn][2], acc[fm][fn][3]};
            float4 s1 = {acc[fm][fn][4], acc[fm][fn][5], acc[fm][fn][6], acc[fm][fn][7]};
            *(float4*)p = s0;
            *((float4*)p + 1) = s1;
        }
    }
}

// ---------------- K3b: reduce split-K -> kv_ext (B-frag layout) ----------------
__global__ __launch_bounds__(256) void la_kv_reduce(const float* __restrict__ kvp,
                                                    unsigned short* __restrict__ kvef) {
    long i = (long)blockIdx.x * 256 + threadIdx.x;   // over 64*128*96
    long bh = i / (VPAD * HEAD_DIM);
    long rem = i - bh * (VPAD * HEAD_DIM);
    int n = (int)(rem / HEAD_DIM), m = (int)(rem % HEAD_DIM);
    float s = 0.0f;
#pragma unroll
    for (int ks = 0; ks < KSPLIT; ++ks)
        s += kvp[(bh * KSPLIT + ks) * (VPAD * HEAD_DIM) + rem];
    kvef[bh * KVE_PER_BH + bfrag_off(m, n, 3)] = f2bf(s);
}

// ---------------- K4: readout GEMM + normalize ----------------
__global__ __launch_bounds__(256) void la_readout(const unsigned short* __restrict__ qff,
                                                  const unsigned short* __restrict__ kvef,
                                                  unsigned short* __restrict__ attnf) {
    __shared__ float norm[128];
    int tid = threadIdx.x, lane = tid & 31, w = tid >> 5;
    int wy = w >> 2, wx = w & 3;
    int m0 = blockIdx.x * 128, bh = blockIdx.y;
    int b = bh >> 3, h = bh & 7;
    const unsigned short* A  = qff + (long)bh * QF_PER_BH;
    const unsigned short* Bk = kvef + (long)bh * KVE_PER_BH;
    v8f acc[4][2];
    v8f z = {0.f, 0.f, 0.f, 0.f, 0.f, 0.f, 0.f, 0.f};
#pragma unroll
    for (int i = 0; i < 4; ++i)
#pragma unroll
        for (int j = 0; j < 2; ++j) acc[i][j] = z;

#pragma unroll
    for (int kt = 0; kt < 3; ++kt) {
        v16bf bfr[2];
#pragma unroll
        for (int fn = 0; fn < 2; ++fn) {
            int nt = wx * 2 + fn;
            bfr[fn] = ldfragg(Bk, ((long)nt * 3 + kt) * 512 + lane * 16);
        }
#pragma unroll
        for (int fm = 0; fm < 4; ++fm) {
            int mt = (m0 >> 4) + wy * 4 + fm;
            v16bf af = ldfragg(A, ((long)mt * 3 + kt) * 512 + lane * 16);
#pragma unroll
            for (int fn = 0; fn < 2; ++fn) acc[fm][fn] = wmma_bf16(af, bfr[fn], acc[fm][fn]);
        }
    }
    // stage normalizer column (global col 96 -> wx==3, fn==0, lane%16==0)
    int hi8 = (lane >> 4) * 8;
    if (wx == 3 && (lane & 15) == 0) {
#pragma unroll
        for (int fm = 0; fm < 4; ++fm)
#pragma unroll
            for (int r = 0; r < 8; ++r)
                norm[wy * 64 + fm * 16 + r + hi8] = acc[fm][0][r];
    }
    __syncthreads();
    int rowbase16 = ((b * SEQ_L + m0) >> 4) + wy * 4;   // global row >> 4 (per fm +fm)
#pragma unroll
    for (int fn = 0; fn < 2; ++fn) {
        int n = wx * 32 + fn * 16 + (lane & 15);
        if (n < HEAD_DIM) {
            int col = h * HEAD_DIM + n;
            int half, e;
            a_halfe(col & 31, half, e);
#pragma unroll
            for (int fm = 0; fm < 4; ++fm) {
                long base = ((long)(rowbase16 + fm) * 24 + (col >> 5)) * 512 +
                            (long)(half * 16 + hi8) * 16 + e;
#pragma unroll
                for (int r = 0; r < 8; ++r) {
                    int ml = wy * 64 + fm * 16 + r + hi8;
                    float den = fmaxf(norm[ml], 1e-6f);
                    attnf[base + r * 16] = f2bf(acc[fm][fn][r] / den);
                }
            }
        }
    }
}

// ---------------- K5: output projection + residual, LDS-free ----------------
__global__ __launch_bounds__(256) void la_gemm_out(const unsigned short* __restrict__ attnf,
                                                   const unsigned short* __restrict__ wof,
                                                   const float* __restrict__ x,
                                                   float* __restrict__ out) {
    int tid = threadIdx.x, lane = tid & 31, w = tid >> 5;
    int wy = w >> 2, wx = w & 3;
    int m0 = blockIdx.y * 128, n0 = blockIdx.x * 128;
    v8f acc[4][2];
    v8f z = {0.f, 0.f, 0.f, 0.f, 0.f, 0.f, 0.f, 0.f};
#pragma unroll
    for (int i = 0; i < 4; ++i)
#pragma unroll
        for (int j = 0; j < 2; ++j) acc[i][j] = z;

    for (int kt = 0; kt < 24; ++kt) {
        v16bf bfr[2];
#pragma unroll
        for (int fn = 0; fn < 2; ++fn) {
            int nt = (n0 >> 4) + wx * 2 + fn;
            bfr[fn] = ldfragg(wof, ((long)nt * 24 + kt) * 512 + lane * 16);
        }
        if (kt + 1 < 24)
            __builtin_prefetch(
                attnf + (((long)(m0 >> 4) + wy * 4) * 24 + kt + 1) * 512 + lane * 16, 0, 1);
#pragma unroll
        for (int fm = 0; fm < 4; ++fm) {
            int mt = (m0 >> 4) + wy * 4 + fm;
            v16bf af = ldfragg(attnf, ((long)mt * 24 + kt) * 512 + lane * 16);
#pragma unroll
            for (int fn = 0; fn < 2; ++fn) acc[fm][fn] = wmma_bf16(af, bfr[fn], acc[fm][fn]);
        }
    }
    int hi8 = (lane >> 4) * 8;
#pragma unroll
    for (int fm = 0; fm < 4; ++fm)
#pragma unroll
        for (int fn = 0; fn < 2; ++fn)
#pragma unroll
            for (int r = 0; r < 8; ++r) {
                long m = m0 + wy * 64 + fm * 16 + r + hi8;
                int n = n0 + wx * 32 + fn * 16 + (lane & 15);
                out[m * D_MODEL + n] = acc[fm][fn][r] + x[m * D_MODEL + n];
            }
}

// ---------------- launcher ----------------
extern "C" void kernel_launch(void* const* d_in, const int* in_sizes, int n_in,
                              void* d_out, int out_size, void* d_ws, size_t ws_size,
                              hipStream_t stream) {
    const float* x     = (const float*)d_in[0];
    const float* gamma = (const float*)d_in[1];
    const float* beta  = (const float*)d_in[2];
    const float* wqkv  = (const float*)d_in[3];
    const float* wout  = (const float*)d_in[4];
    float* out = (float*)d_out;

    char* ws = (char*)d_ws;
    size_t off = 0;
    auto take = [&](size_t bytes) {
        size_t o = off;
        off += (bytes + 255) & ~(size_t)255;
        return o;
    };
    unsigned short* wqkv_bf = (unsigned short*)(ws + take((size_t)D_MODEL * 3 * D_MODEL * 2));
    unsigned short* wout_bf = (unsigned short*)(ws + take((size_t)D_MODEL * D_MODEL * 2));
    unsigned short* xn      = (unsigned short*)(ws + take((size_t)ROWS_TOTAL * D_MODEL * 2));
    unsigned short* qff     = (unsigned short*)(ws + take((size_t)BH * QF_PER_BH * 2));
    unsigned short* ktf     = (unsigned short*)(ws + take((size_t)BH * KT_PER_BH * 2));
    unsigned short* vpf     = (unsigned short*)(ws + take((size_t)BH * VP_PER_BH * 2));
    float*          kvp     = (float*)(ws + take((size_t)BH * KSPLIT * HEAD_DIM * VPAD * 4));
    unsigned short* kvef    = (unsigned short*)(ws + take((size_t)BH * KVE_PER_BH * 2));
    unsigned short* attnf   = (unsigned short*)(ws + take((size_t)ROWS_TOTAL * D_MODEL * 2));

    {   // K0: weight conversion + fragment tiling
        long total = (long)D_MODEL * 3 * D_MODEL + (long)D_MODEL * D_MODEL;
        int blocks = (int)((total + 255) / 256);
        la_wconv<<<blocks, 256, 0, stream>>>(wqkv, wout, wqkv_bf, wout_bf);
    }
    // K1: layernorm
    la_layernorm<<<ROWS_TOTAL / 8, 256, 0, stream>>>(x, gamma, beta, xn);
    // K2: QKV GEMM (N tiles 18, M tiles 256)
    la_gemm_qkv<<<dim3(18, 256), 256, 0, stream>>>(xn, wqkv_bf, qff, ktf, vpf);
    // K3a: KV split-K partials
    la_kv_part<<<dim3(KSPLIT, BH), 256, 0, stream>>>(ktf, vpf, kvp);
    // K3b: reduce
    la_kv_reduce<<<(BH * VPAD * HEAD_DIM) / 256, 256, 0, stream>>>(kvp, kvef);
    // K4: readout (L tiles 32 per head)
    la_readout<<<dim3(SEQ_L / 128, BH), 256, 0, stream>>>(qff, kvef, attnf);
    // K5: output projection + residual (N tiles 6, M tiles 256)
    la_gemm_out<<<dim3(6, 256), 256, 0, stream>>>(attnf, wout_bf, x, out);
}
